// Distribution_74758200754679
// MI455X (gfx1250) — matrix operand order repaired and checked by reading, not compiled, not run on hardware
//
#include <hip/hip_runtime.h>

typedef __attribute__((ext_vector_type(16))) _Float16 v16h;
typedef __attribute__((ext_vector_type(8)))  _Float16 v8h;
typedef __attribute__((ext_vector_type(8)))  float    v8f;
typedef __attribute__((ext_vector_type(4)))  float    v4f;

#define K_D_IN   256
#define K_D_H    128
#define K_SLOPE  0.3f

// One block = 256 threads = 8 waves. Each wave computes a 16-row x 128-col tile
// of h = x @ W1, fused with bias + leaky-relu + dot(W2) + sigmoid + stick-break.
__global__ __launch_bounds__(256) void fused_mlp_stickbreak(
    const float* __restrict__ x,    // [B*N, 256] row-major
    const float* __restrict__ W1,   // [256, 128] row-major
    const float* __restrict__ b1,   // [128]
    const float* __restrict__ W2,   // [128]
    const float* __restrict__ b2,   // [1]
    float* __restrict__ out)        // [B, 9]
{
    // W1 staged as f16, transposed to [n][k], with 16-byte-chunk XOR swizzle:
    // chunk c = k>>3 stored at chunk (c ^ (n & 31)). Makes the 16 lanes of a
    // B-fragment ds_load_b128 hit 16 distinct bank groups. Exactly 64 KB.
    __shared__ _Float16 w1t[K_D_H * K_D_IN];

    const int tid = threadIdx.x;
    #pragma unroll
    for (int i = 0; i < (K_D_IN * K_D_H) / 256; ++i) {
        const int idx = tid + (i << 8);            // coalesced read of W1
        const int k = idx >> 7;                    // 0..255
        const int n = idx & (K_D_H - 1);           // 0..127
        const int kS = (((k >> 3) ^ (n & 31)) << 3) | (k & 7);
        w1t[n * K_D_IN + kS] = (_Float16)W1[idx];
    }
    __syncthreads();

    const int wave   = tid >> 5;
    const int lane   = tid & 31;
    const int halfId = lane >> 4;   // 0: lanes 0-15, 1: lanes 16-31
    const int laneN  = lane & 15;

    const long gw   = (long)blockIdx.x * 8 + wave; // global wave id
    const long row0 = gw * 16;                     // first of 16 rows (b*8+n flat)

    // A-fragment source row for this lane (lanes 0-15 -> M=0..15 with
    // K {0..15}; lanes 16-31 -> M=0..15 with K {16..31}).
    const float* xrow = x + (row0 + laneN) * (long)K_D_IN;
    const int aOff = halfId * 16;

    v8f acc[8];
    #pragma unroll
    for (int t = 0; t < 8; ++t)
        acc[t] = (v8f){0.f, 0.f, 0.f, 0.f, 0.f, 0.f, 0.f, 0.f};

    #pragma unroll
    for (int kk = 0; kk < K_D_IN; kk += 32) {
        // ---- A fragment: 16 halves = one contiguous 16-float run, cvt to f16
        v4f f0 = *(const v4f*)(xrow + kk + aOff);
        v4f f1 = *(const v4f*)(xrow + kk + aOff + 4);
        v4f f2 = *(const v4f*)(xrow + kk + aOff + 8);
        v4f f3 = *(const v4f*)(xrow + kk + aOff + 12);
        v16h afrag;
        #pragma unroll
        for (int i = 0; i < 4; ++i) {
            afrag[i]      = (_Float16)f0[i];
            afrag[4 + i]  = (_Float16)f1[i];
            afrag[8 + i]  = (_Float16)f2[i];
            afrag[12 + i] = (_Float16)f3[i];
        }

        // ---- B fragments from LDS: lane holds col n = 16t+laneN,
        // K = kk + 16*halfId .. +15 (contiguous, 2 swizzled 16B chunks).
        const int kbase = kk + halfId * 16;
        const int c0 = kbase >> 3;
        #pragma unroll
        for (int t = 0; t < 8; ++t) {
            const int n  = t * 16 + laneN;
            const int sw = n & 31;
            const _Float16* rowp = &w1t[n * K_D_IN];
            v8h blo = *(const v8h*)(rowp + (((c0    ) ^ sw) << 3));
            v8h bhi = *(const v8h*)(rowp + (((c0 + 1) ^ sw) << 3));
            v16h bfrag = __builtin_shufflevector(blo, bhi,
                0, 1, 2, 3, 4, 5, 6, 7, 8, 9, 10, 11, 12, 13, 14, 15);
            acc[t] = __builtin_amdgcn_wmma_f32_16x16x32_f16(
                false, afrag, false, bfrag, (short)0, acc[t], false, false);
        }
    }

    // ---- Epilogue. C/D layout: lane (halfId, laneN), VGPR r => element
    // (M = r + 8*halfId, N = 16t + laneN). Fold bias/lrelu/W2 per column.
    float part[8];
    #pragma unroll
    for (int r = 0; r < 8; ++r) part[r] = 0.f;

    #pragma unroll
    for (int t = 0; t < 8; ++t) {
        const int n = t * 16 + laneN;
        const float bb  = b1[n];
        const float w2v = W2[n];
        #pragma unroll
        for (int r = 0; r < 8; ++r) {
            float h = acc[t][r] + bb;
            h = (h >= 0.f) ? h : K_SLOPE * h;
            part[r] += h * w2v;
        }
    }

    // Butterfly-reduce over the 16 lanes of each half-wave (masks 1..8 never
    // cross the halfId boundary under wave32).
    #pragma unroll
    for (int m = 1; m < 16; m <<= 1) {
        #pragma unroll
        for (int r = 0; r < 8; ++r)
            part[r] += __shfl_xor(part[r], m, 32);
    }

    if (laneN == 0) {
        // Half-wave owns one full N=8 group: rows row0 + 8*halfId .. +7,
        // i.e. batch index b = row0/8 + halfId, n = r.
        const float bias2 = b2[0];
        float beta[8];
        #pragma unroll
        for (int r = 0; r < 8; ++r)
            beta[r] = 1.f / (1.f + __expf(-(part[r] + bias2)));

        // Stick-breaking: S[n] = prod_{m>=n} (1-beta[m]);
        // p[0] = S[0]; p[k] = beta[k-1] * S[k]  (k = 1..8).
        float S[9];
        S[8] = 1.f;
        #pragma unroll
        for (int nn = 7; nn >= 0; --nn) S[nn] = (1.f - beta[nn]) * S[nn + 1];

        const long bIdx = (row0 >> 3) + halfId;
        float* o = out + bIdx * 9;
        o[0] = S[0];
        #pragma unroll
        for (int k = 1; k <= 8; ++k) o[k] = beta[k - 1] * S[k];
    }
}

extern "C" void kernel_launch(void* const* d_in, const int* in_sizes, int n_in,
                              void* d_out, int out_size, void* d_ws, size_t ws_size,
                              hipStream_t stream) {
    const float* x  = (const float*)d_in[0];
    const float* W1 = (const float*)d_in[1];
    const float* b1 = (const float*)d_in[2];
    const float* W2 = (const float*)d_in[3];
    const float* b2 = (const float*)d_in[4];
    float* out = (float*)d_out;

    const long rows   = (long)in_sizes[0] / K_D_IN;  // B*N = 524288
    const int  blocks = (int)(rows / 128);           // 128 rows per block
    fused_mlp_stickbreak<<<blocks, 256, 0, stream>>>(x, W1, b1, W2, b2, out);
}
